// ScaledDotProductAttention_22488448762365
// MI455X (gfx1250) — compile-verified, matrix operand order checked by reading, not run on hardware
//
#include <hip/hip_runtime.h>
#include <hip/hip_bf16.h>

typedef __attribute__((ext_vector_type(16))) __bf16 v16bf;
typedef __attribute__((ext_vector_type(8)))  float  v8f;

#define KL_STRIDE 72            // halves per K-tile row (64 + pad): 144B, 16B-aligned
#define VT_STRIDE 40            // halves per V^T row   (32 + pad):  80B, 16B-aligned
#define KL_ELEMS (32 * KL_STRIDE)
#define VT_ELEMS (64 * VT_STRIDE)

// Read a 32-element chunk of a bf16 LDS row into WMMA 16-bit A/B layout:
// element h -> row[base + (h/8)*16 + g*8 + (h%8)]   (two aligned 16B loads)
__device__ __forceinline__ v16bf lds_chunk(const __bf16* row, int g, int base) {
  union { uint4 u[2]; v16bf v; } t;
  t.u[0] = *(const uint4*)(row + base + g * 8);
  t.u[1] = *(const uint4*)(row + base + 16 + g * 8);
  return t.v;
}

struct StageRegs { float4 ka, kb, va, vb; };

__device__ __forceinline__ void stage_load(const float* __restrict__ Kb,
                                           const float* __restrict__ Vb,
                                           int k0, int skey, int sd0, StageRegs& r) {
  const float4* kg = (const float4*)(Kb + (size_t)(k0 + skey) * 64 + sd0);
  const float4* vg = (const float4*)(Vb + (size_t)(k0 + skey) * 64 + sd0);
  r.ka = kg[0]; r.kb = kg[1];
  r.va = vg[0]; r.vb = vg[1];
}

__device__ __forceinline__ void stage_store(__bf16* klb, __bf16* vtb,
                                            int skey, int sd0, const StageRegs& r) {
  union { __bf16 h[8]; uint4 u; } kp;
  kp.h[0] = (__bf16)r.ka.x; kp.h[1] = (__bf16)r.ka.y;
  kp.h[2] = (__bf16)r.ka.z; kp.h[3] = (__bf16)r.ka.w;
  kp.h[4] = (__bf16)r.kb.x; kp.h[5] = (__bf16)r.kb.y;
  kp.h[6] = (__bf16)r.kb.z; kp.h[7] = (__bf16)r.kb.w;
  *(uint4*)&klb[skey * KL_STRIDE + sd0] = kp.u;     // one aligned 16B LDS store
  const float vv[8] = {r.va.x, r.va.y, r.va.z, r.va.w, r.vb.x, r.vb.y, r.vb.z, r.vb.w};
#pragma unroll
  for (int j = 0; j < 8; ++j)                       // transposed scatter: V^T[dim][key]
    vtb[(sd0 + j) * VT_STRIDE + skey] = (__bf16)vv[j];
}

__global__ __launch_bounds__(256) void fa_fwd_bf16_wmma(
    const float* __restrict__ Q, const float* __restrict__ K,
    const float* __restrict__ V, const int* __restrict__ vlen,
    float* __restrict__ O, int B, int Nq, int Nkv) {
  constexpr int D = 64;
  __shared__ __align__(16) __bf16 kl[2][KL_ELEMS];   // K tile, key-major bf16 (double buffer)
  __shared__ __align__(16) __bf16 vt[2][VT_ELEMS];   // V^T tile, dim-major bf16 (double buffer)

  const int tid  = threadIdx.x;
  const int lane = tid & 31;
  const int wave = tid >> 5;
  const int blocksPerB = (Nq >> 4) >> 3;             // 8 query-tiles (128 queries) per block
  const int b  = blockIdx.x / blocksPerB;
  const int qt = (blockIdx.x % blocksPerB) * 8 + wave;
  const int g  = lane >> 4;                          // half-wave group (0/1)
  const int nl = lane & 15;                          // lane within group

  const float* Qb = Q + (size_t)b * Nq  * D;
  const float* Kb = K + (size_t)b * Nkv * D;
  const float* Vb = V + (size_t)b * Nkv * D;
  const int vl = vlen[b];

  // Keys >= vl carry zero softmax weight once any valid key exists -> stop early.
  // vl==0 runs the full masked loop (reproduces the reference's uniform softmax).
  const int kEnd = (vl > 0) ? min((vl + 31) & ~31, Nkv) : Nkv;
  const int T = kEnd >> 5;                           // 32-key tiles (block-uniform)

  // Staging assignment: 32x64 tile = 2048 elems, 8 per thread (one eighth-row chunk)
  const int skey = tid >> 3;
  const int sd0  = (tid & 7) << 3;

  // Q^T pre-scaled by (1/sqrt(D)) * log2(e): scores live in the log2 domain, so
  // softmax weights use raw v_exp_f32 (2^x) with no per-element multiply.
  const int q = qt * 16 + nl;
  const float qscale = 0.125f * 1.44269504088896340736f;
  v16bf bq0, bq1;
  {
    const float* row = Qb + (size_t)q * D;
#pragma unroll
    for (int h = 0; h < 16; ++h) {
      const int d = ((h >> 3) << 4) + (g << 3) + (h & 7);
      bq0[h] = (__bf16)(row[d]      * qscale);
      bq1[h] = (__bf16)(row[d + 32] * qscale);
    }
  }

  float m = -3.0e38f;            // running row max (log2 domain)
  float l = 0.0f;                // running row sum
  v8f o[4] = {};

  // Pipeline prologue: stage tile 0
  StageRegs sr;
  stage_load(Kb, Vb, 0, skey, sd0, sr);
  stage_store(kl[0], vt[0], skey, sd0, sr);
  __syncthreads();

  for (int t = 0; t < T; ++t) {
    const int k0 = t << 5;
    const bool more = (t + 1 < T);
    StageRegs nxt;
    if (more) stage_load(Kb, Vb, k0 + 32, skey, sd0, nxt);   // loads fly under compute

    // ---- S^T = K * Q^T : two 16-key subtiles, each over D=64 (2 k-chunks) ----
    const __bf16* kb_ = kl[t & 1];
    const __bf16* vb_ = vt[t & 1];
    const __bf16* krow0 = kb_ + nl * KL_STRIDE;
    const __bf16* krow1 = kb_ + (16 + nl) * KL_STRIDE;
    v8f st0 = {}, st1 = {};
    st0 = __builtin_amdgcn_wmma_f32_16x16x32_bf16(false, lds_chunk(krow0, g, 0),  false, bq0, (short)0, st0, false, false);
    st1 = __builtin_amdgcn_wmma_f32_16x16x32_bf16(false, lds_chunk(krow1, g, 0),  false, bq0, (short)0, st1, false, false);
    st0 = __builtin_amdgcn_wmma_f32_16x16x32_bf16(false, lds_chunk(krow0, g, 32), false, bq1, (short)0, st0, false, false);
    st1 = __builtin_amdgcn_wmma_f32_16x16x32_bf16(false, lds_chunk(krow1, g, 32), false, bq1, (short)0, st1, false, false);

    // ---- mask only in the boundary tile; C layout: st_s[i] -> key k0+s*16+8g+i ----
    float s0[8], s1[8];
    if (k0 + 32 <= vl) {                           // fully-valid fast path (uniform branch)
#pragma unroll
      for (int i = 0; i < 8; ++i) { s0[i] = st0[i]; s1[i] = st1[i]; }
    } else {                                       // cold: last tile (or vl==0 fallback)
#pragma unroll
      for (int i = 0; i < 8; ++i) {
        const int key0 = k0 + 8 * g + i;
        s0[i] = (key0      < vl) ? st0[i] : -3.0e38f;
        s1[i] = (key0 + 16 < vl) ? st1[i] : -3.0e38f;
      }
    }
    float tmax = -3.0e38f;
#pragma unroll
    for (int i = 0; i < 8; ++i) tmax = fmaxf(tmax, fmaxf(s0[i], s1[i]));
    tmax = fmaxf(tmax, __shfl_xor(tmax, 16));      // partner half-wave: other 16 keys
    const float mn    = fmaxf(m, tmax);
    const float alpha = __builtin_amdgcn_exp2f(m - mn);

    // ---- P^T = 2^(S^T - m); lands directly in B-matrix layout for PV WMMA ----
    float psum = 0.0f;
    v16bf pb;
#pragma unroll
    for (int i = 0; i < 8; ++i) {
      const float p0 = __builtin_amdgcn_exp2f(s0[i] - mn);
      const float p1 = __builtin_amdgcn_exp2f(s1[i] - mn);
      psum += p0 + p1;
      pb[i]     = (__bf16)p0;                      // local key 8g+i
      pb[i + 8] = (__bf16)p1;                      // local key 16+8g+i
    }
    psum += __shfl_xor(psum, 16);
    l = l * alpha + psum;

    if (__any(mn > m)) {                           // rescale only on a new max (uniform branch)
#pragma unroll
      for (int dt = 0; dt < 4; ++dt)
#pragma unroll
        for (int i = 0; i < 8; ++i) o[dt][i] *= alpha;
    }
    m = mn;

    // ---- O^T += V^T * P^T : A = V^T chunk (16 dims x 32 keys) per dim-tile ----
#pragma unroll
    for (int dt = 0; dt < 4; ++dt) {
      const __bf16* vrow = vb_ + (dt * 16 + nl) * VT_STRIDE;
      o[dt] = __builtin_amdgcn_wmma_f32_16x16x32_bf16(false, lds_chunk(vrow, g, 0), false, pb, (short)0, o[dt], false, false);
    }

    if (more) stage_store(kl[(t + 1) & 1], vt[(t + 1) & 1], skey, sd0, nxt);
    __syncthreads();                               // next tile visible; this tile's reads done
  }

  // ---- normalize & store: o[dt][i] -> dim = dt*16 + 8g + i, query = q ----
  const float inv_l = 1.0f / l;
  float* Ob = O + ((size_t)b * Nq + q) * D;
#pragma unroll
  for (int dt = 0; dt < 4; ++dt) {
    const int dbase = dt * 16 + 8 * g;
    float4 lo = {o[dt][0] * inv_l, o[dt][1] * inv_l, o[dt][2] * inv_l, o[dt][3] * inv_l};
    float4 hi = {o[dt][4] * inv_l, o[dt][5] * inv_l, o[dt][6] * inv_l, o[dt][7] * inv_l};
    *(float4*)(Ob + dbase)     = lo;               // 16B-aligned contiguous stores
    *(float4*)(Ob + dbase + 4) = hi;
  }
}

extern "C" void kernel_launch(void* const* d_in, const int* in_sizes, int n_in,
                              void* d_out, int out_size, void* d_ws, size_t ws_size,
                              hipStream_t stream) {
  const float* Q   = (const float*)d_in[0];
  const float* K   = (const float*)d_in[1];
  const float* V   = (const float*)d_in[2];
  const int*  vlen = (const int*) d_in[3];
  float* O = (float*)d_out;

  const int D   = 64;
  const int B   = in_sizes[3];
  const int Nq  = in_sizes[0] / (B * D);
  const int Nkv = in_sizes[1] / (B * D);

  const int blocks = B * (Nq / 128);               // 8 waves x 16 queries per block
  fa_fwd_bf16_wmma<<<blocks, 256, 0, stream>>>(Q, K, V, vlen, O, B, Nq, Nkv);
}
